// GlobalAttention_73461120630936
// MI455X (gfx1250) — compile-verified, matrix operand order checked by reading, not run on hardware
//
#include <hip/hip_runtime.h>
#include <hip/hip_bf16.h>

// ---------------------------------------------------------------------------
// Global attention (B=8, H=W=32, C=768, heads=12, hd=64) for gfx1250 (MI455X).
// All matmuls on v_wmma_f32_16x16x32_f16 (wave32 WMMA), f32 accumulate.
// Software-pipelined (double-buffered fragment loads ahead of MMAs); k-loops
// unrolled x2 so the double buffers ping-pong without register copies.
// ---------------------------------------------------------------------------

typedef __attribute__((ext_vector_type(4)))  _Float16 v4h;
typedef __attribute__((ext_vector_type(8)))  _Float16 v8h;
typedef __attribute__((ext_vector_type(16))) _Float16 v16h;
typedef __attribute__((ext_vector_type(8)))  float    v8f;

__device__ __forceinline__ v8f wmma_f32_f16(v16h a, v16h b, v8f c) {
  return __builtin_amdgcn_wmma_f32_16x16x32_f16(
      false, a, false, b, (short)0, c, false, false);
}

// 16x32 f16 fragment in CDNA5 A/B VGPR layout from a row-major matrix:
//   lane l: row = row0 + (l&15); halves 0-7  = cols k0 + (l>>4)*8 + 0..7
//                                halves 8-15 = cols k0 + 16 + (l>>4)*8 + 0..7
// Two contiguous 16-byte loads per lane.  Also serves B when the source is
// B^T row-major (lane = B column = source row).
__device__ __forceinline__ v16h load_frag16(const _Float16* __restrict__ base,
                                            int ld, int row0, int k0) {
  int lane = threadIdx.x & 31;
  int m = lane & 15;
  int g = lane >> 4;
  const _Float16* p = base + (size_t)(row0 + m) * (size_t)ld + (size_t)(k0 + g * 8);
  v8h lo = *(const v8h*)(p);
  v8h hi = *(const v8h*)(p + 16);
  return __builtin_shufflevector(lo, hi, 0,1,2,3,4,5,6,7,8,9,10,11,12,13,14,15);
}

// ---------------------------------------------------------------------------
// f32 -> f16 conversion (vectorized x4)
// ---------------------------------------------------------------------------
__global__ void __launch_bounds__(256) cvt_f32_to_f16(const float* __restrict__ in,
                                                      _Float16* __restrict__ out,
                                                      int n4) {
  int i = blockIdx.x * blockDim.x + threadIdx.x;
  if (i < n4) {
    float4 v = ((const float4*)in)[i];
    v4h o = { (_Float16)v.x, (_Float16)v.y, (_Float16)v.z, (_Float16)v.w };
    ((v4h*)out)[i] = o;
  }
}

// ---------------------------------------------------------------------------
// QKV GEMM: C[8192,2304] = Xh[8192,768] * Wqkv[2304,768]^T + b
// Wave tile: 32x64 (2 A frags x 4 B frags -> 8 accumulators).  Loads for
// k-step k+32 are issued before the WMMAs of k-step k; unroll x2 ping-pongs
// the buffers.  B-frag loads issue first, A-frags last (consumed first).
// Epilogue scatters q (x 1/8), k as (96,1024,64), v transposed (96,64,1024).
// ---------------------------------------------------------------------------
__global__ void __launch_bounds__(256) qkv_gemm(const _Float16* __restrict__ xh,
                                                const _Float16* __restrict__ wh,
                                                const float* __restrict__ bias,
                                                _Float16* __restrict__ qb,
                                                _Float16* __restrict__ kb,
                                                _Float16* __restrict__ vtb) {
  const int K = 768;
  const int nStrips = 2304 / 64;                 // 36
  int wid = blockIdx.x * 8 + (threadIdx.x >> 5); // 9216 waves
  int mBase = (wid / nStrips) * 32;
  int nBase = (wid % nStrips) * 64;

  v8f acc0[4] = {}, acc1[4] = {};
  v16h b[4];
#pragma unroll
  for (int t = 0; t < 4; ++t) b[t] = load_frag16(wh, K, nBase + t * 16, 0);
  v16h a0 = load_frag16(xh, K, mBase,      0);
  v16h a1 = load_frag16(xh, K, mBase + 16, 0);

#pragma unroll 2
  for (int k0 = 0; k0 < K; k0 += 32) {
    int kn = k0 + 32; if (kn >= K) kn = 0;       // last prefetch wraps (dead)
    v16h nb[4];
#pragma unroll
    for (int t = 0; t < 4; ++t) nb[t] = load_frag16(wh, K, nBase + t * 16, kn);
    v16h na0 = load_frag16(xh, K, mBase,      kn);
    v16h na1 = load_frag16(xh, K, mBase + 16, kn);

#pragma unroll
    for (int t = 0; t < 4; ++t) acc0[t] = wmma_f32_f16(a0, b[t], acc0[t]);
#pragma unroll
    for (int t = 0; t < 4; ++t) acc1[t] = wmma_f32_f16(a1, b[t], acc1[t]);

    a0 = na0; a1 = na1;
#pragma unroll
    for (int t = 0; t < 4; ++t) b[t] = nb[t];
  }

  int lane = threadIdx.x & 31;
  int cn = lane & 15, hi = lane >> 4;
  int t3  = nBase / 768;            // 0=q 1=k 2=v (uniform per wave)
  int h   = (nBase % 768) / 64;     // head (uniform per wave)
  const float qscale = 0.125f;      // hd^-0.5

#pragma unroll
  for (int s = 0; s < 2; ++s) {
    int mb = mBase + s * 16;
#pragma unroll
    for (int t = 0; t < 4; ++t) {
      int n  = nBase + t * 16 + cn;
      float bv = bias[n];
      int d  = t * 16 + cn;
#pragma unroll
      for (int e = 0; e < 8; ++e) {
        float val = (s ? acc1[t][e] : acc0[t][e]) + bv;
        int r   = mb + e + hi * 8;  // token row
        int bb  = r >> 10;          // batch
        int pos = r & 1023;         // position
        size_t gh = (size_t)(bb * 12 + h);
        if (t3 == 0) {
          qb[(gh * 1024 + pos) * 64 + d] = (_Float16)(val * qscale);
        } else if (t3 == 1) {
          kb[(gh * 1024 + pos) * 64 + d] = (_Float16)val;
        } else {
          vtb[(gh * 64 + d) * 1024 + pos] = (_Float16)val;  // transposed V
        }
      }
    }
  }
}

// ---------------------------------------------------------------------------
// Flash attention: one wave per (head, 16-query tile).  Pipelined: next
// chunk's K fragments and current chunk's V fragments are issued before the
// softmax VALU block and LDS P-transpose, hiding their latency behind ALU.
// ---------------------------------------------------------------------------
__global__ void __launch_bounds__(128) attn_kernel(const _Float16* __restrict__ qb,
                                                   const _Float16* __restrict__ kb,
                                                   const _Float16* __restrict__ vtb,
                                                   _Float16* __restrict__ ob) {
  const int L = 1024, HD = 64;
  __shared__ _Float16 pLds[4][16 * 32];          // 1KB P tile per wave

  int wslot = threadIdx.x >> 5;
  int wid = blockIdx.x * 4 + wslot;              // 6144 waves
  int g  = wid >> 6;                             // head 0..95
  int qt = wid & 63;                             // query tile

  const _Float16* qh = qb  + (size_t)g * L * HD;
  const _Float16* kh = kb  + (size_t)g * L * HD;
  const _Float16* vh = vtb + (size_t)g * HD * L; // (64, 1024)
  _Float16* pbuf = &pLds[wslot][0];

  int lane = threadIdx.x & 31;
  int cn = lane & 15, hi = lane >> 4;

  v16h aq0 = load_frag16(qh, HD, qt * 16, 0);    // q pre-scaled by 1/8
  v16h aq1 = load_frag16(qh, HD, qt * 16, 32);

  v8f o0 = {}, o1 = {}, o2 = {}, o3 = {};
  float mrow[8], lrow[8];
#pragma unroll
  for (int e = 0; e < 8; ++e) { mrow[e] = -1e30f; lrow[e] = 0.0f; }

  // preload K fragments for first chunk
  v16h kf0 = load_frag16(kh, HD,  0, 0);
  v16h kf1 = load_frag16(kh, HD,  0, 32);
  v16h kf2 = load_frag16(kh, HD, 16, 0);
  v16h kf3 = load_frag16(kh, HD, 16, 32);

#pragma unroll 2
  for (int kc = 0; kc < L; kc += 32) {
    // ---- scores for 32 keys ----
    v8f s0 = {}, s1 = {};
    s0 = wmma_f32_f16(aq0, kf0, s0);
    s0 = wmma_f32_f16(aq1, kf1, s0);
    s1 = wmma_f32_f16(aq0, kf2, s1);
    s1 = wmma_f32_f16(aq1, kf3, s1);

    // ---- issue next-chunk K loads and current V loads (overlap softmax) ----
    int kn = (kc + 32) & (L - 1);                // wraps on last iter (dead)
    kf0 = load_frag16(kh, HD, kn,      0);
    kf1 = load_frag16(kh, HD, kn,      32);
    kf2 = load_frag16(kh, HD, kn + 16, 0);
    kf3 = load_frag16(kh, HD, kn + 16, 32);
    v16h vf0 = load_frag16(vh, L,  0, kc);
    v16h vf1 = load_frag16(vh, L, 16, kc);
    v16h vf2 = load_frag16(vh, L, 32, kc);
    v16h vf3 = load_frag16(vh, L, 48, kc);
    if (kc + 64 < L)
      __builtin_prefetch(kh + (size_t)(kc + 64) * HD, 0, 1);

    // ---- online softmax (rows in 16-lane halves, one row per VGPR) ----
    float alpha[8];
#pragma unroll
    for (int e = 0; e < 8; ++e) {
      float mx = fmaxf(s0[e], s1[e]);
      mx = fmaxf(mx, __shfl_xor(mx, 1, 32));
      mx = fmaxf(mx, __shfl_xor(mx, 2, 32));
      mx = fmaxf(mx, __shfl_xor(mx, 4, 32));
      mx = fmaxf(mx, __shfl_xor(mx, 8, 32));
      float mnew = fmaxf(mrow[e], mx);
      float a  = __expf(mrow[e] - mnew);
      float p0 = __expf(s0[e] - mnew);
      float p1 = __expf(s1[e] - mnew);
      float rs = p0 + p1;
      rs += __shfl_xor(rs, 1, 32);
      rs += __shfl_xor(rs, 2, 32);
      rs += __shfl_xor(rs, 4, 32);
      rs += __shfl_xor(rs, 8, 32);
      lrow[e] = lrow[e] * a + rs;
      mrow[e] = mnew;
      alpha[e] = a;
      int row = e + hi * 8;
      pbuf[row * 32 + cn]      = (_Float16)p0;   // row-major 16x32 P tile
      pbuf[row * 32 + 16 + cn] = (_Float16)p1;
    }
    __builtin_amdgcn_wave_barrier();             // keep DS store->load order
    v16h ap = load_frag16(pbuf, 32, 0, 0);       // P as A-fragment
    __builtin_amdgcn_wave_barrier();

    // ---- rescale running output, accumulate P(16x32) * V(32x64) ----
#pragma unroll
    for (int e = 0; e < 8; ++e) {
      o0[e] *= alpha[e]; o1[e] *= alpha[e];
      o2[e] *= alpha[e]; o3[e] *= alpha[e];
    }
    o0 = wmma_f32_f16(ap, vf0, o0);
    o1 = wmma_f32_f16(ap, vf1, o1);
    o2 = wmma_f32_f16(ap, vf2, o2);
    o3 = wmma_f32_f16(ap, vf3, o3);
  }

  // ---- normalize and write (token-major (8192, 768) f16) ----
  int b = g / 12, h = g % 12;
#pragma unroll
  for (int e = 0; e < 8; ++e) {
    float rinv = 1.0f / lrow[e];
    int pos = qt * 16 + e + hi * 8;
    size_t rowoff = ((size_t)b * 1024 + pos) * 768 + (size_t)h * 64;
    ob[rowoff +  0 + cn] = (_Float16)(o0[e] * rinv);
    ob[rowoff + 16 + cn] = (_Float16)(o1[e] * rinv);
    ob[rowoff + 32 + cn] = (_Float16)(o2[e] * rinv);
    ob[rowoff + 48 + cn] = (_Float16)(o3[e] * rinv);
  }
}

// ---------------------------------------------------------------------------
// Output projection: out[8192,768] = Ah[8192,768] * Wproj[768,768]^T + b, f32
// Same pipelined 32x64 wave tile as qkv_gemm.
// ---------------------------------------------------------------------------
__global__ void __launch_bounds__(256) proj_gemm(const _Float16* __restrict__ ah,
                                                 const _Float16* __restrict__ wh,
                                                 const float* __restrict__ bias,
                                                 float* __restrict__ out) {
  const int N = 768, K = 768;
  int wid = blockIdx.x * 8 + (threadIdx.x >> 5); // 3072 waves
  int mBase = (wid / 12) * 32;
  int nBase = (wid % 12) * 64;

  v8f acc0[4] = {}, acc1[4] = {};
  v16h b[4];
#pragma unroll
  for (int t = 0; t < 4; ++t) b[t] = load_frag16(wh, K, nBase + t * 16, 0);
  v16h a0 = load_frag16(ah, K, mBase,      0);
  v16h a1 = load_frag16(ah, K, mBase + 16, 0);

#pragma unroll 2
  for (int k0 = 0; k0 < K; k0 += 32) {
    int kn = k0 + 32; if (kn >= K) kn = 0;
    v16h nb[4];
#pragma unroll
    for (int t = 0; t < 4; ++t) nb[t] = load_frag16(wh, K, nBase + t * 16, kn);
    v16h na0 = load_frag16(ah, K, mBase,      kn);
    v16h na1 = load_frag16(ah, K, mBase + 16, kn);

#pragma unroll
    for (int t = 0; t < 4; ++t) acc0[t] = wmma_f32_f16(a0, b[t], acc0[t]);
#pragma unroll
    for (int t = 0; t < 4; ++t) acc1[t] = wmma_f32_f16(a1, b[t], acc1[t]);

    a0 = na0; a1 = na1;
#pragma unroll
    for (int t = 0; t < 4; ++t) b[t] = nb[t];
  }

  int lane = threadIdx.x & 31;
  int cn = lane & 15, hi = lane >> 4;
#pragma unroll
  for (int s = 0; s < 2; ++s) {
    int mb = mBase + s * 16;
#pragma unroll
    for (int t = 0; t < 4; ++t) {
      int n = nBase + t * 16 + cn;
      float bv = bias[n];
#pragma unroll
      for (int e = 0; e < 8; ++e) {
        int r = mb + e + hi * 8;
        out[(size_t)r * N + n] = (s ? acc1[t][e] : acc0[t][e]) + bv;
      }
    }
  }
}

// ---------------------------------------------------------------------------
// Host launcher
// ---------------------------------------------------------------------------
extern "C" void kernel_launch(void* const* d_in, const int* in_sizes, int n_in,
                              void* d_out, int out_size, void* d_ws, size_t ws_size,
                              hipStream_t stream) {
  (void)in_sizes; (void)n_in; (void)out_size; (void)ws_size;
  const float* x      = (const float*)d_in[0];  // (8,32,32,768)
  const float* qkv_w  = (const float*)d_in[1];  // (2304,768)
  const float* qkv_b  = (const float*)d_in[2];  // (2304,)
  const float* proj_w = (const float*)d_in[3];  // (768,768)
  const float* proj_b = (const float*)d_in[4];  // (768,)
  float* out = (float*)d_out;                   // (8,32,32,768) f32

  _Float16* ws = (_Float16*)d_ws;
  size_t off = 0;
  _Float16* xh     = ws + off; off += (size_t)8192 * 768;
  _Float16* wqkvh  = ws + off; off += (size_t)2304 * 768;
  _Float16* wprojh = ws + off; off += (size_t)768  * 768;
  _Float16* qbuf   = ws + off; off += (size_t)96 * 1024 * 64;
  _Float16* kbuf   = ws + off; off += (size_t)96 * 1024 * 64;
  _Float16* vtbuf  = ws + off; off += (size_t)96 * 64 * 1024;
  _Float16* obuf   = ws + off; off += (size_t)8192 * 768;

  cvt_f32_to_f16<<<(8192 * 768 / 4 + 255) / 256, 256, 0, stream>>>(x, xh, 8192 * 768 / 4);
  cvt_f32_to_f16<<<(2304 * 768 / 4 + 255) / 256, 256, 0, stream>>>(qkv_w, wqkvh, 2304 * 768 / 4);
  cvt_f32_to_f16<<<(768 * 768 / 4 + 255) / 256, 256, 0, stream>>>(proj_w, wprojh, 768 * 768 / 4);

  // (8192/32)*(2304/64) = 9216 waves, 8 per block
  qkv_gemm<<<9216 / 8, 256, 0, stream>>>(xh, wqkvh, qkv_b, qbuf, kbuf, vtbuf);

  // 96 heads * 64 query tiles = 6144 waves, 4 per block
  attn_kernel<<<6144 / 4, 128, 0, stream>>>(qbuf, kbuf, vtbuf, obuf);

  // (8192/32)*(768/64) = 3072 waves, 8 per block
  proj_gemm<<<3072 / 8, 256, 0, stream>>>(obuf, wprojh, proj_b, out);
}
